// TransformerBlock_31387620999462
// MI455X (gfx1250) — matrix operand order repaired and checked by reading, not compiled, not run on hardware
//
#include <hip/hip_runtime.h>
#include <hip/hip_bf16.h>
#include <math.h>

typedef unsigned short bf16s;  // bf16 storage type
typedef __attribute__((ext_vector_type(16))) __bf16 v16bf;
typedef __attribute__((ext_vector_type(8)))  float  v8f;

union FragU { v16bf v; unsigned u[8]; };

// float -> bf16 (round to nearest even)
__device__ __forceinline__ bf16s f2b(float f) {
  unsigned u = __float_as_uint(f);
  u += 0x7FFFu + ((u >> 16) & 1u);
  return (bf16s)(u >> 16);
}

// ---- CDNA5 async global->LDS copy (ASYNCcnt), 16B per lane ----
__device__ __forceinline__ void async_b128(const void* lds_dst, const void* gsrc) {
  unsigned lds_off = (unsigned)(uintptr_t)lds_dst;          // low 32b of generic LDS addr
  unsigned long long ga = (unsigned long long)(uintptr_t)gsrc;
  asm volatile("global_load_async_to_lds_b128 %0, %1, off"
               :
               : "v"(lds_off), "v"(ga)
               : "memory");
}
__device__ __forceinline__ void async_wait0() {
  asm volatile("s_wait_asynccnt 0x0" ::: "memory");
}

// A-matrix fragment, 16x32 bf16 (ISA 7.12.2): lane holds row m=lane&15,
// VGPR v holds k-pair at ((v<4)?0:16) + ((lane>=16)?8:0) + 2*(v&3)
__device__ __forceinline__ v16bf load_frag_a(const bf16s* __restrict__ base, int ld) {
  const int lane = threadIdx.x & 31;
  const int m  = lane & 15;
  const int kh = (lane & 16) >> 1;  // 0 or 8
  FragU f;
#pragma unroll
  for (int v = 0; v < 8; ++v) {
    const int k = ((v & 4) << 2) + kh + ((v & 3) << 1);
    f.u[v] = *(const unsigned*)(base + m * ld + k);
  }
  return f.v;
}

// B-matrix fragment, 32x16 bf16: lane holds col n=lane&15,
// VGPR v holds k-pair at ((v<4)?0:16) + ((lane>=16)?8:0) + 2*(v&3).  Expects B stored [n][k].
__device__ __forceinline__ v16bf load_frag_b(const bf16s* __restrict__ base, int ld) {
  const int lane = threadIdx.x & 31;
  const int n  = lane & 15;
  const int kh = (lane & 16) >> 1;  // 0 or 8
  FragU f;
#pragma unroll
  for (int v = 0; v < 8; ++v) {
    const int k = ((v & 4) << 2) + kh + ((v & 3) << 1);
    f.u[v] = *(const unsigned*)(base + n * ld + k);
  }
  return f.v;
}

__device__ __forceinline__ v8f wmma_bf(v16bf a, v16bf b, v8f c) {
  return __builtin_amdgcn_wmma_f32_16x16x32_bf16(false, a, false, b, (short)0, c, false, false);
}

__device__ __forceinline__ float rmax16(float x) {
  x = fmaxf(x, __shfl_xor(x, 1, 32));
  x = fmaxf(x, __shfl_xor(x, 2, 32));
  x = fmaxf(x, __shfl_xor(x, 4, 32));
  x = fmaxf(x, __shfl_xor(x, 8, 32));
  return x;
}
__device__ __forceinline__ float rsum16(float x) {
  x += __shfl_xor(x, 1, 32);
  x += __shfl_xor(x, 2, 32);
  x += __shfl_xor(x, 4, 32);
  x += __shfl_xor(x, 8, 32);
  return x;
}

// ---------------- z = x + positional encoding ----------------
__global__ __launch_bounds__(256) void pe_kernel(const float* __restrict__ x,
                                                 float* __restrict__ zf,
                                                 bf16s* __restrict__ zb) {
  const int idx = blockIdx.x * 256 + threadIdx.x;     // (sin,cos) pair index
  const int d2 = idx & 255;
  const int s  = (idx >> 8) & 2047;
  const float div = __expf((float)(2 * d2) * (-9.210340372f / 512.0f));
  const float ang = (float)s * div;
  const size_t base = (size_t)idx * 2;
  const float ze = x[base]     + __sinf(ang);
  const float zo = x[base + 1] + __cosf(ang);
  zf[base] = ze; zf[base + 1] = zo;
  zb[base] = f2b(ze); zb[base + 1] = f2b(zo);
}

// ---------------- W[512][512] f32 -> Wt[n][k] bf16 ----------------
__global__ __launch_bounds__(256) void wt_kernel(const float* __restrict__ W,
                                                 bf16s* __restrict__ Wt) {
  const int idx = blockIdx.x * 256 + threadIdx.x;
  const int n = idx & 511;
  const int k = idx >> 9;
  Wt[(size_t)n * 512 + k] = f2b(W[idx]);
}

// ---------------- bf16 GEMM, 128x128 tile, double-buffered async LDS ----------------
// C[M][N] = epi(A[M][K] @ Bt[N][K]^T + bias).  8 waves, each 32(M) x 64(N).
// Epilogue is specialized at compile time -> branch-free unrolled code.
template <bool HasRes, bool HasF, bool HasB, bool Leaky>
__global__ __launch_bounds__(256) void gemm_bf16_kernel(
    const bf16s* __restrict__ A, const bf16s* __restrict__ Bt,
    const float* __restrict__ bias, const float* __restrict__ Res,
    float* __restrict__ outF, bf16s* __restrict__ outB,
    int M, int N, int K, float scale)
{
  __shared__ __align__(16) bf16s la[2][128][32];
  __shared__ __align__(16) bf16s lb[2][128][32];

  const int m0 = blockIdx.x * 128;
  const int n0 = blockIdx.y * 128;
  const int tid = threadIdx.x;
  const int wave = tid >> 5;
  const int lane = tid & 31;
  const int wm = (wave & 3) << 5;   // 0/32/64/96
  const int wn = (wave >> 2) << 6;  // 0/64

  const v8f vzero = {};
  v8f acc[2][4];
#pragma unroll
  for (int i = 0; i < 2; ++i)
#pragma unroll
    for (int j = 0; j < 4; ++j) acc[i][j] = vzero;

  const int lr = tid >> 2;          // 0..63
  const int lc = (tid & 3) << 3;    // 0/8/16/24 (16B)

  // prologue: async-stage tile k=0 into buffer 0
  async_b128(&la[0][lr][lc],      A  + (size_t)(m0 + lr)      * K + lc);
  async_b128(&la[0][lr + 64][lc], A  + (size_t)(m0 + lr + 64) * K + lc);
  async_b128(&lb[0][lr][lc],      Bt + (size_t)(n0 + lr)      * K + lc);
  async_b128(&lb[0][lr + 64][lc], Bt + (size_t)(n0 + lr + 64) * K + lc);
  async_wait0();
  __syncthreads();

  int cur = 0;
  for (int k0 = 0; k0 < K; k0 += 32) {
    const bool more = (k0 + 32) < K;
    if (more) {  // async DMA of next tile overlaps WMMA on current tile
      const int kn = k0 + 32;
      const int nb = cur ^ 1;
      async_b128(&la[nb][lr][lc],      A  + (size_t)(m0 + lr)      * K + kn + lc);
      async_b128(&la[nb][lr + 64][lc], A  + (size_t)(m0 + lr + 64) * K + kn + lc);
      async_b128(&lb[nb][lr][lc],      Bt + (size_t)(n0 + lr)      * K + kn + lc);
      async_b128(&lb[nb][lr + 64][lc], Bt + (size_t)(n0 + lr + 64) * K + kn + lc);
      if (kn + 32 < K)
        __builtin_prefetch(A + (size_t)(m0 + lr) * K + kn + 32 + lc, 0, 1);
    }

    v16bf a0 = load_frag_a(&la[cur][wm][0], 32);
    v16bf a1 = load_frag_a(&la[cur][wm + 16][0], 32);
    v16bf bfr[4];
#pragma unroll
    for (int j = 0; j < 4; ++j) bfr[j] = load_frag_b(&lb[cur][wn + j * 16][0], 32);
#pragma unroll
    for (int j = 0; j < 4; ++j) {
      acc[0][j] = wmma_bf(a0, bfr[j], acc[0][j]);
      acc[1][j] = wmma_bf(a1, bfr[j], acc[1][j]);
    }

    if (more) async_wait0();
    __syncthreads();
    cur ^= 1;
  }

  // branch-free epilogue
  const int hf = (lane >> 4) << 3;  // 0 or 8
  const int cn = lane & 15;
  const int colBase = n0 + wn + cn;

  float bv[4];
#pragma unroll
  for (int j = 0; j < 4; ++j) bv[j] = bias[colBase + j * 16];

#pragma unroll
  for (int i = 0; i < 2; ++i) {
#pragma unroll
    for (int r = 0; r < 8; ++r) {
      const int row = m0 + wm + i * 16 + r + hf;
      const size_t rb = (size_t)row * N + colBase;
      float res[4];
      if (HasRes) {
#pragma unroll
        for (int j = 0; j < 4; ++j) res[j] = Res[rb + j * 16];
      }
#pragma unroll
      for (int j = 0; j < 4; ++j) {
        float v = (acc[i][j][r] + bv[j]) * scale;
        if (HasRes) v += res[j];
        if (Leaky)  v = (v > 0.0f) ? v : 0.01f * v;
        if (HasF) outF[rb + j * 16] = v;
        if (HasB) outB[rb + j * 16] = f2b(v);
      }
    }
  }
}

// ---------------- flash attention ----------------
// grid.x = S/128, grid.y = B*H.  8 waves x 16 query rows each.
// Q is pre-scaled by 1/sqrt(Dh) in its GEMM epilogue.
__global__ __launch_bounds__(256) void attn_kernel(
    const bf16s* __restrict__ Q, const bf16s* __restrict__ Kmat,
    const bf16s* __restrict__ Vmat, bf16s* __restrict__ O)
{
  __shared__ __align__(16) bf16s lk[64][64];       // [t][d]  (serves QK^T B-frag)
  __shared__ __align__(16) bf16s lv[64][64];       // [d][t]  (serves PV  B-frag)
  __shared__ __align__(16) bf16s lp[8][16][64];    // per-wave P tile

  const int bh = blockIdx.y;
  const int bb = bh >> 3, hh = bh & 7;
  const int m0 = blockIdx.x * 128;
  const int tid = threadIdx.x, wave = tid >> 5, lane = tid & 31;
  const int hc = hh << 6;
  const size_t rowBase = (size_t)bb * 2048 + m0 + wave * 16;

  const bf16s* qbase = Q + rowBase * 512 + hc;
  v16bf qa0 = load_frag_a(qbase, 512);
  v16bf qa1 = load_frag_a(qbase + 32, 512);

  float mrow[8], lrow[8], alpha[8];
#pragma unroll
  for (int r = 0; r < 8; ++r) { mrow[r] = -1e30f; lrow[r] = 0.0f; }
  v8f o0 = {}, o1 = {}, o2 = {}, o3 = {};

  const int hf = (lane >> 4) << 3;
  const int cn = lane & 15;
  const int lr = tid >> 3;         // 0..31
  const int lc = (tid & 7) << 3;   // 0..56

  for (int t0 = 0; t0 < 2048; t0 += 64) {
    __syncthreads();
    const bf16s* kb = Kmat + ((size_t)bb * 2048 + t0) * 512 + hc;
    const bf16s* vb = Vmat + ((size_t)bb * 2048 + t0) * 512 + hc;
    // K tile: async DMA straight into LDS (no VGPR round-trip)
    async_b128(&lk[lr][lc],      kb + (size_t)lr * 512 + lc);
    async_b128(&lk[lr + 32][lc], kb + (size_t)(lr + 32) * 512 + lc);
    // V tile: manual load + transpose store into lv[d][t]
    uint4 vv0 = *(const uint4*)(vb + (size_t)lr * 512 + lc);
    uint4 vv1 = *(const uint4*)(vb + (size_t)(lr + 32) * 512 + lc);
    const bf16s* p0 = (const bf16s*)&vv0;
    const bf16s* p1 = (const bf16s*)&vv1;
#pragma unroll
    for (int e = 0; e < 8; ++e) {
      lv[lc + e][lr]      = p0[e];
      lv[lc + e][lr + 32] = p1[e];
    }
    async_wait0();
    __syncthreads();

    // scores: 16x64 tile = 4 n-subtiles x 2 k-steps
    v8f s0 = {}, s1 = {}, s2 = {}, s3 = {};
    s0 = wmma_bf(qa0, load_frag_b(&lk[0][0],   64), s0);
    s0 = wmma_bf(qa1, load_frag_b(&lk[0][32],  64), s0);
    s1 = wmma_bf(qa0, load_frag_b(&lk[16][0],  64), s1);
    s1 = wmma_bf(qa1, load_frag_b(&lk[16][32], 64), s1);
    s2 = wmma_bf(qa0, load_frag_b(&lk[32][0],  64), s2);
    s2 = wmma_bf(qa1, load_frag_b(&lk[32][32], 64), s2);
    s3 = wmma_bf(qa0, load_frag_b(&lk[48][0],  64), s3);
    s3 = wmma_bf(qa1, load_frag_b(&lk[48][32], 64), s3);

    // online softmax (row = r + hf, distributed over 16 lanes)
#pragma unroll
    for (int r = 0; r < 8; ++r) {
      float mx = fmaxf(fmaxf(s0[r], s1[r]), fmaxf(s2[r], s3[r]));
      mx = rmax16(mx);
      const float mn = fmaxf(mrow[r], mx);
      alpha[r] = __expf(mrow[r] - mn);
      mrow[r] = mn;
      const float p0v = __expf(s0[r] - mn);
      const float p1v = __expf(s1[r] - mn);
      const float p2v = __expf(s2[r] - mn);
      const float p3v = __expf(s3[r] - mn);
      lp[wave][r + hf][cn]      = f2b(p0v);
      lp[wave][r + hf][16 + cn] = f2b(p1v);
      lp[wave][r + hf][32 + cn] = f2b(p2v);
      lp[wave][r + hf][48 + cn] = f2b(p3v);
      lrow[r] = lrow[r] * alpha[r] + rsum16(p0v + p1v + p2v + p3v);
      o0[r] *= alpha[r]; o1[r] *= alpha[r]; o2[r] *= alpha[r]; o3[r] *= alpha[r];
    }
    __syncthreads();

    // O += P @ V
    v16bf pa0 = load_frag_a(&lp[wave][0][0],  64);
    v16bf pa1 = load_frag_a(&lp[wave][0][32], 64);
    o0 = wmma_bf(pa0, load_frag_b(&lv[0][0],   64), o0);
    o0 = wmma_bf(pa1, load_frag_b(&lv[0][32],  64), o0);
    o1 = wmma_bf(pa0, load_frag_b(&lv[16][0],  64), o1);
    o1 = wmma_bf(pa1, load_frag_b(&lv[16][32], 64), o1);
    o2 = wmma_bf(pa0, load_frag_b(&lv[32][0],  64), o2);
    o2 = wmma_bf(pa1, load_frag_b(&lv[32][32], 64), o2);
    o3 = wmma_bf(pa0, load_frag_b(&lv[48][0],  64), o3);
    o3 = wmma_bf(pa1, load_frag_b(&lv[48][32], 64), o3);
  }

  bf16s* obase = O + rowBase * 512 + hc;
#pragma unroll
  for (int r = 0; r < 8; ++r) {
    const float inv = 1.0f / lrow[r];
    obase[(size_t)(r + hf) * 512 + cn]      = f2b(o0[r] * inv);
    obase[(size_t)(r + hf) * 512 + 16 + cn] = f2b(o1[r] * inv);
    obase[(size_t)(r + hf) * 512 + 32 + cn] = f2b(o2[r] * inv);
    obase[(size_t)(r + hf) * 512 + 48 + cn] = f2b(o3[r] * inv);
  }
}

extern "C" void kernel_launch(void* const* d_in, const int* in_sizes, int n_in,
                              void* d_out, int out_size, void* d_ws, size_t ws_size,
                              hipStream_t stream) {
  const float* x  = (const float*)d_in[0];
  const float* Wq = (const float*)d_in[1];  const float* bq = (const float*)d_in[2];
  const float* Wk = (const float*)d_in[3];  const float* bk = (const float*)d_in[4];
  const float* Wv = (const float*)d_in[5];  const float* bv = (const float*)d_in[6];
  const float* Wo = (const float*)d_in[7];  const float* bo = (const float*)d_in[8];
  const float* W1 = (const float*)d_in[9];  const float* b1 = (const float*)d_in[10];
  const float* W2 = (const float*)d_in[11]; const float* b2 = (const float*)d_in[12];

  const int M = 4 * 2048, N = 512, K = 512;
  char* ws = (char*)d_ws;
  size_t off = 0;
  auto carve = [&](size_t bytes) { void* p = ws + off; off += (bytes + 255) & ~(size_t)255; return p; };

  float* zf  = (float*)carve((size_t)M * N * 4);   // z (residual, f32)
  bf16s* zb  = (bf16s*)carve((size_t)M * N * 2);   // z bf16
  bf16s* qb  = (bf16s*)carve((size_t)M * N * 2);   // Q (pre-scaled)
  bf16s* kb  = (bf16s*)carve((size_t)M * N * 2);   // K
  bf16s* vb  = (bf16s*)carve((size_t)M * N * 2);   // V
  bf16s* ab  = (bf16s*)carve((size_t)M * N * 2);   // attention out
  float* z2f = (float*)carve((size_t)M * N * 4);   // z2 f32
  bf16s* z2b = (bf16s*)carve((size_t)M * N * 2);   // z2 bf16
  bf16s* fb  = (bf16s*)carve((size_t)M * N * 2);   // ffn hidden
  bf16s* wqt = (bf16s*)carve((size_t)K * N * 2);
  bf16s* wkt = (bf16s*)carve((size_t)K * N * 2);
  bf16s* wvt = (bf16s*)carve((size_t)K * N * 2);
  bf16s* wot = (bf16s*)carve((size_t)K * N * 2);
  bf16s* w1t = (bf16s*)carve((size_t)K * N * 2);
  bf16s* w2t = (bf16s*)carve((size_t)K * N * 2);

  pe_kernel<<<8192, 256, 0, stream>>>(x, zf, zb);
  wt_kernel<<<1024, 256, 0, stream>>>(Wq, wqt);
  wt_kernel<<<1024, 256, 0, stream>>>(Wk, wkt);
  wt_kernel<<<1024, 256, 0, stream>>>(Wv, wvt);
  wt_kernel<<<1024, 256, 0, stream>>>(Wo, wot);
  wt_kernel<<<1024, 256, 0, stream>>>(W1, w1t);
  wt_kernel<<<1024, 256, 0, stream>>>(W2, w2t);

  dim3 gg(M / 128, N / 128);
  // Q/K/V projections (Q carries the 1/sqrt(64) scale) -> bf16 only
  gemm_bf16_kernel<false, false, true, false><<<gg, 256, 0, stream>>>(
      zb, wqt, bq, nullptr, nullptr, qb, M, N, K, 0.125f);
  gemm_bf16_kernel<false, false, true, false><<<gg, 256, 0, stream>>>(
      zb, wkt, bk, nullptr, nullptr, kb, M, N, K, 1.0f);
  gemm_bf16_kernel<false, false, true, false><<<gg, 256, 0, stream>>>(
      zb, wvt, bv, nullptr, nullptr, vb, M, N, K, 1.0f);

  dim3 ga(2048 / 128, 4 * 8);
  attn_kernel<<<ga, 256, 0, stream>>>(qb, kb, vb, ab);

  // z2 = z + attn @ Wo + bo  (residual, f32 + bf16 outputs)
  gemm_bf16_kernel<true, true, true, false><<<gg, 256, 0, stream>>>(
      ab, wot, bo, zf, z2f, z2b, M, N, K, 1.0f);
  // f = leaky(z2 @ W1 + b1)  -> bf16 only
  gemm_bf16_kernel<false, false, true, true><<<gg, 256, 0, stream>>>(
      z2b, w1t, b1, nullptr, nullptr, fb, M, N, K, 1.0f);
  // out = leaky(f @ W2 + b2) -> f32 (d_out)
  gemm_bf16_kernel<false, true, false, true><<<gg, 256, 0, stream>>>(
      fb, w2t, b2, nullptr, (float*)d_out, nullptr, M, N, K, 1.0f);
}
